// NConvUNet_18614388261295
// MI455X (gfx1250) — matrix operand order repaired and checked by reading, not compiled, not run on hardware
//
#include <hip/hip_runtime.h>
#include <hip/hip_bf16.h>

typedef __attribute__((ext_vector_type(16))) _Float16 v16h;
typedef __attribute__((ext_vector_type(8)))  float    v8f;

#define EPSF 1e-20f

// ---------------------------------------------------------------------------
// Weight prep: per-Cout weight sums (for cout = denom / sum(w)) and packing
// fp32 OIHW weights into a per-lane 16-bit WMMA A-operand layout so the conv
// kernel loads one contiguous v16h (32B) per lane per (tap, chunk, mtile).
// A k order used here (16x32 f16, same as B staging):  lane L holds M = L%16;  element e -> K =
//     16*(L>>4) + e
// ---------------------------------------------------------------------------
__global__ void k_wsum(const float* __restrict__ w, float* __restrict__ out, int per) {
  int co = threadIdx.x;
  if (co >= 32) return;
  float s = 0.f;
  for (int j = 0; j < per; ++j) s += w[co * per + j];
  out[co] = s;
}

__global__ void k_pack(const float* __restrict__ w, _Float16* __restrict__ dst,
                       int Cin, int K, int nchunk) {
  int i = blockIdx.x * 256 + threadIdx.x;
  int total = K * K * nchunk * 2 * 32 * 16;
  if (i >= total) return;
  int e     = i & 15;
  int lane  = (i >> 4) & 31;
  int m     = (i >> 9) & 1;
  int rest  = i >> 10;
  int chunk = rest % nchunk;
  int t     = rest / nchunk;
  int co = m * 16 + (lane & 15);
  int kk = 16 * (lane >> 4) + e;
  int ci = chunk * 32 + kk;
  int ky = t / K, kx = t % K;
  dst[i] = (_Float16)w[((co * Cin + ci) * K + ky) * K + kx];
}

// ---------------------------------------------------------------------------
// Stage 1: nconv with w1 (Cin=1, 5x5). Tiny FLOPs -> scalar fp32, reads the
// raw fp32 input (ch0 = data, ch1 = confidence), writes f16 x/c tensors.
// ---------------------------------------------------------------------------
__global__ void k_nconv1(const float* __restrict__ in, const float* __restrict__ w1,
                         const float* __restrict__ b1, const float* __restrict__ ws1,
                         _Float16* __restrict__ xO, _Float16* __restrict__ cO,
                         int H, int W) {
  int i = blockIdx.x * 256 + threadIdx.x;
  int HW = H * W;
  int total = 2 * 32 * HW;
  if (i >= total) return;
  int p  = i % HW;
  int co = (i / HW) % 32;
  int n  = i / (32 * HW);
  int y = p / W, x = p % W;
  const float* x0 = in + (size_t)n * 2 * HW;
  const float* c0 = x0 + HW;
  float nom = 0.f, den = 0.f;
  for (int ky = 0; ky < 5; ++ky) {
    int yy = y + ky - 2;
    if ((unsigned)yy >= (unsigned)H) continue;
    for (int kx = 0; kx < 5; ++kx) {
      int xx = x + kx - 2;
      if ((unsigned)xx >= (unsigned)W) continue;
      float xv = x0[yy * W + xx], cv = c0[yy * W + xx];
      float wv = w1[co * 25 + ky * 5 + kx];
      nom += xv * cv * wv;
      den += cv * wv;
    }
  }
  float out  = nom / (den + EPSF) + b1[co];
  float cout = den / ws1[co];
  size_t o = ((size_t)n * 32 + co) * HW + p;
  xO[o] = (_Float16)out;
  cO[o] = (_Float16)cout;
}

// ---------------------------------------------------------------------------
// 2x2 stride-2 max-pool on c, gather x at argmax (first-max tie break, window
// order (r0,c0),(r0,c1),(r1,c0),(r1,c1) to match the reference reshape).
// ---------------------------------------------------------------------------
__global__ void k_pool(const _Float16* __restrict__ xI, const _Float16* __restrict__ cI,
                       _Float16* __restrict__ xO, _Float16* __restrict__ cO,
                       int H, int W) {
  int Ho = H >> 1, Wo = W >> 1, HWo = Ho * Wo;
  int i = blockIdx.x * 256 + threadIdx.x;
  int total = 2 * 32 * HWo;
  if (i >= total) return;
  int p  = i % HWo;
  int ch = (i / HWo) % 32;
  int n  = i / (32 * HWo);
  int yo = p / Wo, xo = p % Wo;
  size_t ib = ((size_t)n * 32 + ch) * H * W + (size_t)(2 * yo) * W + 2 * xo;
  float c00 = (float)cI[ib],     c01 = (float)cI[ib + 1];
  float c10 = (float)cI[ib + W], c11 = (float)cI[ib + W + 1];
  int best = 0; float bc = c00;
  if (c01 > bc) { bc = c01; best = 1; }
  if (c10 > bc) { bc = c10; best = 2; }
  if (c11 > bc) { bc = c11; best = 3; }
  float bx = (best == 0) ? (float)xI[ib]
           : (best == 1) ? (float)xI[ib + 1]
           : (best == 2) ? (float)xI[ib + W]
                         : (float)xI[ib + W + 1];
  size_t o = ((size_t)n * 32 + ch) * HWo + p;
  xO[o] = (_Float16)bx;
  cO[o] = (_Float16)(bc * 0.25f);
}

// ---------------------------------------------------------------------------
// Final nconv with w7 (1x1, 32->1). Output fp32 straight to d_out.
// ---------------------------------------------------------------------------
__global__ void k_nconv7(const _Float16* __restrict__ xI, const _Float16* __restrict__ cI,
                         const float* __restrict__ w7, const float* __restrict__ b7,
                         float* __restrict__ out, int H, int W) {
  int HW = H * W;
  int i = blockIdx.x * 256 + threadIdx.x;
  int total = 2 * HW;
  if (i >= total) return;
  int p = i % HW;
  int n = i / HW;
  float nom = 0.f, den = 0.f;
  for (int ci = 0; ci < 32; ++ci) {
    size_t o = ((size_t)n * 32 + ci) * HW + p;
    float xv = (float)xI[o], cv = (float)cI[o];
    float wv = w7[ci];
    nom += xv * cv * wv;
    den += cv * wv;
  }
  out[(size_t)n * HW + p] = nom / (den + EPSF) + b7[0];
}

// ---------------------------------------------------------------------------
// Core: implicit-GEMM normalized conv via v_wmma_f32_16x16x32_f16 with LDS
// tile staging.
//
// One 256-thread block = 8 wave32; each wave owns a 16-pixel row strip:
// M = 32 outs (two 16-row A tiles), N = 16 px, K = 32 channels per chunk per
// filter tap.  The (16+K-1)x(8+K-1) halo tile (x*c and c, per chunk) is
// staged into LDS ONCE in B-fragment layout [chunk][ci_half][pixel][16ch]:
// a lane's 16-channel B fragment is one contiguous, 32B-aligned v16h row, so
// the tap loop is pure ds_load_b128 + v_wmma with zero divergence (padding
// zero-filled at stage time; EXEC all-ones as WMMA requires).  x*c is formed
// once at stage time and reused by all ~K*K taps.
//
// NCHUNK=2 reads channels 0-31 from source A and 32-63 from source B (fused
// concat); per-source shift fuses nearest-neighbor 2x upsampling into the
// staging gather.  B k order used here (32x16 f16, same as A packing): lane L holds N = L%16,
// element e -> K = 16*(L>>4)+e.  C/D: VGPR r -> M = r + 8*(L>>4), N = L%16.
// ---------------------------------------------------------------------------
template <int KS, int NCHUNK>
__global__ __launch_bounds__(256) void k_nconv_wmma(
    const _Float16* __restrict__ xA, const _Float16* __restrict__ cA, int shiftA,
    const _Float16* __restrict__ xB, const _Float16* __restrict__ cB, int shiftB,
    const _Float16* __restrict__ wpack, const float* __restrict__ wsum,
    const float* __restrict__ bias,
    _Float16* __restrict__ xO, _Float16* __restrict__ cO,
    int H, int W) {
  constexpr int PAD  = KS / 2;
  constexpr int TH   = 8 + KS - 1;    // tile rows incl. halo
  constexpr int TW   = 16 + KS - 1;   // tile cols incl. halo
  constexpr int NPIX = TH * TW;

  __shared__ _Float16 sP[NCHUNK][2][NPIX][16];  // x*c, B-fragment layout
  __shared__ _Float16 sC[NCHUNK][2][NPIX][16];  // c

  const int tid  = threadIdx.x;
  const int lane = tid & 31;
  const int wv   = tid >> 5;
  const int hi   = lane >> 4;
  const int nlo  = lane & 15;
  const int n    = blockIdx.z;
  const int y0   = blockIdx.y * 8;
  const int x0   = blockIdx.x * 16;

  // ---- stage halo tile(s) into LDS (x fastest -> coalesced global reads) ---
  for (int s = tid; s < NCHUNK * 32 * NPIX; s += 256) {
    int px = s % TW;
    int t1 = s / TW;
    int py = t1 % TH;
    int t2 = t1 / TH;
    int c  = t2 & 31;
    int chunk = t2 >> 5;
    const _Float16* sx = (chunk == 0) ? xA : xB;
    const _Float16* sc = (chunk == 0) ? cA : cB;
    const int sh = (chunk == 0) ? shiftA : shiftB;
    const int Hs = H >> sh, Ws = W >> sh;
    const int gy = y0 + py - PAD;
    const int gx = x0 + px - PAD;
    _Float16 xv = (_Float16)0.f, cv = (_Float16)0.f;
    if ((unsigned)gy < (unsigned)H && (unsigned)gx < (unsigned)W) {
      size_t off = ((size_t)n * 32 + c) * (size_t)(Hs * Ws) +
                   (size_t)(gy >> sh) * Ws + (size_t)(gx >> sh);
      xv = sx[off];
      cv = sc[off];
    }
    int pix = py * TW + px;
    sP[chunk][c >> 4][pix][c & 15] = xv * cv;
    sC[chunk][c >> 4][pix][c & 15] = cv;
  }
  __syncthreads();

  // ---- tap loop: branch-free ds_load_b128 + WMMA ---------------------------
  v8f aX0 = {}, aX1 = {}, aC0 = {}, aC1 = {};
  for (int ky = 0; ky < KS; ++ky) {
#pragma unroll
    for (int kx = 0; kx < KS; ++kx) {
      const int t   = ky * KS + kx;
      const int pix = (wv + ky) * TW + (nlo + kx);
#pragma unroll
      for (int chunk = 0; chunk < NCHUNK; ++chunk) {
        v16h bx = *(const v16h*)&sP[chunk][hi][pix][0];
        v16h bc = *(const v16h*)&sC[chunk][hi][pix][0];
        const v16h* Ap = ((const v16h*)wpack) + (size_t)(t * NCHUNK + chunk) * 64;
        v16h a0 = Ap[lane];        // M rows 0..15
        v16h a1 = Ap[32 + lane];   // M rows 16..31
        aX0 = __builtin_amdgcn_wmma_f32_16x16x32_f16(false, a0, false, bx, (short)0, aX0, false, false);
        aC0 = __builtin_amdgcn_wmma_f32_16x16x32_f16(false, a0, false, bc, (short)0, aC0, false, false);
        aX1 = __builtin_amdgcn_wmma_f32_16x16x32_f16(false, a1, false, bx, (short)0, aX1, false, false);
        aC1 = __builtin_amdgcn_wmma_f32_16x16x32_f16(false, a1, false, bc, (short)0, aC1, false, false);
      }
    }
  }

  // ---- epilogue: normalize, bias, conf propagate ---------------------------
  const int y  = y0 + wv;
  const int xq = x0 + nlo;
  const size_t HW = (size_t)H * W;
#pragma unroll
  for (int r = 0; r < 8; ++r) {
    int m0 = r + 8 * hi;
    int m1 = m0 + 16;
    float d0 = aC0[r];
    float o0 = aX0[r] / (d0 + EPSF) + bias[m0];
    float q0 = d0 / wsum[m0];
    size_t off0 = ((size_t)n * 32 + m0) * HW + (size_t)y * W + xq;
    xO[off0] = (_Float16)o0;
    cO[off0] = (_Float16)q0;
    float d1 = aC1[r];
    float o1 = aX1[r] / (d1 + EPSF) + bias[m1];
    float q1 = d1 / wsum[m1];
    size_t off1 = ((size_t)n * 32 + m1) * HW + (size_t)y * W + xq;
    xO[off1] = (_Float16)o1;
    cO[off1] = (_Float16)q1;
  }
}

// ---------------------------------------------------------------------------
// Host-side orchestration.
// ---------------------------------------------------------------------------
extern "C" void kernel_launch(void* const* d_in, const int* in_sizes, int n_in,
                              void* d_out, int out_size, void* d_ws, size_t ws_size,
                              hipStream_t stream) {
  (void)in_sizes; (void)n_in; (void)out_size; (void)ws_size;
  const float* x  = (const float*)d_in[0];
  const float* w1 = (const float*)d_in[1];  const float* b1 = (const float*)d_in[2];
  const float* w2 = (const float*)d_in[3];  const float* b2 = (const float*)d_in[4];
  const float* w3 = (const float*)d_in[5];  const float* b3 = (const float*)d_in[6];
  const float* w4 = (const float*)d_in[7];  const float* b4 = (const float*)d_in[8];
  const float* w5 = (const float*)d_in[9];  const float* b5 = (const float*)d_in[10];
  const float* w6 = (const float*)d_in[11]; const float* b6 = (const float*)d_in[12];
  const float* w7 = (const float*)d_in[13]; const float* b7 = (const float*)d_in[14];
  float* out = (float*)d_out;

  // workspace carve-out (all intermediates f16)
  char* base = (char*)d_ws;
  size_t cur = 0;
  auto alloc = [&](size_t bytes) -> void* {
    void* p = base + cur;
    cur = (cur + bytes + 255) & ~(size_t)255;
    return p;
  };
  const size_t T512 = (size_t)2 * 32 * 512 * 512 * sizeof(_Float16);
  const size_t T256 = T512 / 4, T128 = T512 / 16, T64 = T512 / 64;

  _Float16 *A512x = (_Float16*)alloc(T512), *A512c = (_Float16*)alloc(T512);
  _Float16 *B512x = (_Float16*)alloc(T512), *B512c = (_Float16*)alloc(T512);
  _Float16 *C512x = (_Float16*)alloc(T512), *C512c = (_Float16*)alloc(T512); // x1,c1
  _Float16 *A256x = (_Float16*)alloc(T256), *A256c = (_Float16*)alloc(T256);
  _Float16 *B256x = (_Float16*)alloc(T256), *B256c = (_Float16*)alloc(T256);
  _Float16 *C256x = (_Float16*)alloc(T256), *C256c = (_Float16*)alloc(T256); // x2_ds
  _Float16 *A128x = (_Float16*)alloc(T128), *A128c = (_Float16*)alloc(T128);
  _Float16 *B128x = (_Float16*)alloc(T128), *B128c = (_Float16*)alloc(T128); // x3_ds
  _Float16 *C128x = (_Float16*)alloc(T128), *C128c = (_Float16*)alloc(T128); // x34_ds
  _Float16 *A64x  = (_Float16*)alloc(T64),  *A64c  = (_Float16*)alloc(T64);
  _Float16 *B64x  = (_Float16*)alloc(T64),  *B64c  = (_Float16*)alloc(T64);  // x4_ds
  const size_t PK5 = (size_t)25 * 1 * 2 * 32 * 16 * sizeof(_Float16);
  const size_t PK3 = (size_t)9  * 2 * 2 * 32 * 16 * sizeof(_Float16);
  _Float16* wp2 = (_Float16*)alloc(PK5);
  _Float16* wp3 = (_Float16*)alloc(PK5);
  _Float16* wp4 = (_Float16*)alloc(PK3);
  _Float16* wp5 = (_Float16*)alloc(PK3);
  _Float16* wp6 = (_Float16*)alloc(PK3);
  float* wsums = (float*)alloc(6 * 32 * sizeof(float)); // w1..w6

  // weight sums + packing
  k_wsum<<<1, 32, 0, stream>>>(w1, wsums + 0,   25);
  k_wsum<<<1, 32, 0, stream>>>(w2, wsums + 32,  800);
  k_wsum<<<1, 32, 0, stream>>>(w3, wsums + 64,  800);
  k_wsum<<<1, 32, 0, stream>>>(w4, wsums + 96,  576);
  k_wsum<<<1, 32, 0, stream>>>(w5, wsums + 128, 576);
  k_wsum<<<1, 32, 0, stream>>>(w6, wsums + 160, 576);
  k_pack<<<(25600 + 255) / 256, 256, 0, stream>>>(w2, wp2, 32, 5, 1);
  k_pack<<<(25600 + 255) / 256, 256, 0, stream>>>(w3, wp3, 32, 5, 1);
  k_pack<<<(18432 + 255) / 256, 256, 0, stream>>>(w4, wp4, 64, 3, 2);
  k_pack<<<(18432 + 255) / 256, 256, 0, stream>>>(w5, wp5, 64, 3, 2);
  k_pack<<<(18432 + 255) / 256, 256, 0, stream>>>(w6, wp6, 64, 3, 2);

  auto cgrid = [](int H, int W) { return dim3((unsigned)(W / 16), (unsigned)(H / 8), 2); };
  const dim3 blk(256);

  // encoder
  k_nconv1<<<(2 * 32 * 512 * 512) / 256, blk, 0, stream>>>(x, w1, b1, wsums + 0, A512x, A512c, 512, 512);
  k_nconv_wmma<5, 1><<<cgrid(512, 512), blk, 0, stream>>>(A512x, A512c, 0, nullptr, nullptr, 0,
      wp2, wsums + 32, b2, B512x, B512c, 512, 512);
  k_nconv_wmma<5, 1><<<cgrid(512, 512), blk, 0, stream>>>(B512x, B512c, 0, nullptr, nullptr, 0,
      wp3, wsums + 64, b3, C512x, C512c, 512, 512);                                  // x1,c1
  k_pool<<<(2 * 32 * 256 * 256) / 256, blk, 0, stream>>>(C512x, C512c, A256x, A256c, 512, 512);
  k_nconv_wmma<5, 1><<<cgrid(256, 256), blk, 0, stream>>>(A256x, A256c, 0, nullptr, nullptr, 0,
      wp2, wsums + 32, b2, B256x, B256c, 256, 256);
  k_nconv_wmma<5, 1><<<cgrid(256, 256), blk, 0, stream>>>(B256x, B256c, 0, nullptr, nullptr, 0,
      wp3, wsums + 64, b3, C256x, C256c, 256, 256);                                  // x2_ds
  k_pool<<<(2 * 32 * 128 * 128) / 256, blk, 0, stream>>>(C256x, C256c, A128x, A128c, 256, 256);
  k_nconv_wmma<5, 1><<<cgrid(128, 128), blk, 0, stream>>>(A128x, A128c, 0, nullptr, nullptr, 0,
      wp2, wsums + 32, b2, B128x, B128c, 128, 128);                                  // x3_ds
  k_pool<<<(2 * 32 * 64 * 64) / 256, blk, 0, stream>>>(B128x, B128c, A64x, A64c, 128, 128);
  k_nconv_wmma<5, 1><<<cgrid(64, 64), blk, 0, stream>>>(A64x, A64c, 0, nullptr, nullptr, 0,
      wp2, wsums + 32, b2, B64x, B64c, 64, 64);                                      // x4_ds

  // decoder (concat + 2x nearest upsample fused into staging of the low-res chunk)
  k_nconv_wmma<3, 2><<<cgrid(128, 128), blk, 0, stream>>>(B128x, B128c, 0, B64x, B64c, 1,
      wp4, wsums + 96, b4, C128x, C128c, 128, 128);                                  // x34_ds
  k_nconv_wmma<3, 2><<<cgrid(256, 256), blk, 0, stream>>>(C256x, C256c, 0, C128x, C128c, 1,
      wp5, wsums + 128, b5, A256x, A256c, 256, 256);                                 // x23_ds
  k_nconv_wmma<3, 2><<<cgrid(512, 512), blk, 0, stream>>>(A256x, A256c, 1, C512x, C512c, 0,
      wp6, wsums + 160, b6, A512x, A512c, 512, 512);                                 // xout pre-1x1
  k_nconv7<<<(2 * 512 * 512) / 256, blk, 0, stream>>>(A512x, A512c, w7, b7, out, 512, 512);
}
